// HMMLanguageModel_47588237640158
// MI455X (gfx1250) — matrix operand order repaired and checked by reading, not compiled, not run on hardware
//
#include <hip/hip_runtime.h>
#include <hip/hip_bf16.h>

typedef __attribute__((ext_vector_type(16))) _Float16 v16h;
typedef __attribute__((ext_vector_type(8)))  float    v8f;

#define HMM_S     128   // num states
#define HMM_B     256   // batch
#define HMM_VOCAB 32000
#define BCOLS     32    // batch columns per workgroup
#define EPITCH    136   // LDS pitch for exp-buffer rows (272B = 17*16B: aligned + conflict-free)
#define FPITCH    132   // LDS pitch for f rows

#define LOG2E 1.4426950408889634f
#define LN2   0.6931471805599453f

__device__ __forceinline__ float fast_exp(float x) {
  return __builtin_amdgcn_exp2f(x * LOG2E);     // v_exp_f32
}
__device__ __forceinline__ float fast_log(float x) {
  return __builtin_amdgcn_logf(x) * LN2;        // v_log_f32
}

// ---------------------------------------------------------------------------
// Kernel A: per-state logsumexp over the vocab axis: lse[s] = LSE_v table[v,s]
// ---------------------------------------------------------------------------
__global__ __launch_bounds__(256) void hmm_col_lse(const float* __restrict__ table,
                                                   float* __restrict__ lse) {
  const int s = blockIdx.x;
  const int tid = threadIdx.x;
  __shared__ float red[256];
  float m = -INFINITY;
  for (int v = tid; v < HMM_VOCAB; v += 256) m = fmaxf(m, table[v * HMM_S + s]);
  red[tid] = m; __syncthreads();
  for (int off = 128; off > 0; off >>= 1) {
    if (tid < off) red[tid] = fmaxf(red[tid], red[tid + off]);
    __syncthreads();
  }
  m = red[0]; __syncthreads();
  float sum = 0.f;
  for (int v = tid; v < HMM_VOCAB; v += 256) sum += fast_exp(table[v * HMM_S + s] - m);
  red[tid] = sum; __syncthreads();
  for (int off = 128; off > 0; off >>= 1) {
    if (tid < off) red[tid] += red[tid + off];
    __syncthreads();
  }
  if (tid == 0) lse[s] = fast_log(red[0]) + m;
}

// ---------------------------------------------------------------------------
// Kernel B: row softmax of transition -> f16 probability matrix Th[j][k]
// ---------------------------------------------------------------------------
__global__ __launch_bounds__(128) void hmm_row_softmax(const float* __restrict__ trans,
                                                       _Float16* __restrict__ Th) {
  const int j = blockIdx.x, k = threadIdx.x;
  __shared__ float red[128];
  const float x = trans[j * HMM_S + k];
  red[k] = x; __syncthreads();
  for (int off = 64; off > 0; off >>= 1) {
    if (k < off) red[k] = fmaxf(red[k], red[k + off]);
    __syncthreads();
  }
  const float m = red[0]; __syncthreads();
  const float e = fast_exp(x - m);
  red[k] = e; __syncthreads();
  for (int off = 64; off > 0; off >>= 1) {
    if (k < off) red[k] += red[k + off];
    __syncthreads();
  }
  Th[j * HMM_S + k] = (_Float16)(e / red[0]);
}

// ---------------------------------------------------------------------------
// Kernel C: persistent scaled-forward recursion, 2 barriers per timestep.
//  WG = 32 batch columns; wave w owns j-tile [16w,16w+16), two 16x16 n-tiles.
//  step: E=exp(f-c) (f16,LDS) -> 8x v_wmma_f32_16x16x32_f16 (B-frags preloaded)
//        -> f=log(acc)+c+em_t ; next step's scaling max produced in-register
//        via __shfl_xor(16) and one pmax2[b][wave] store per column.
// ---------------------------------------------------------------------------
__global__ __launch_bounds__(256) void hmm_forward(const int* __restrict__ sent,
                                                   const float* __restrict__ table,
                                                   const float* __restrict__ lse,
                                                   const _Float16* __restrict__ Th,
                                                   float* __restrict__ partial,
                                                   int L) {
  __shared__ float    f_lds[BCOLS][FPITCH];
  __shared__ _Float16 E_lds[BCOLS][EPITCH];
  __shared__ float    c_lds[BCOLS];
  __shared__ float    pmax2[BCOLS][8];   // per-(column, wave) partial maxima
  __shared__ float    pred[BCOLS][8];
  __shared__ float    lse_lds[HMM_S];
  __shared__ float    colppl[BCOLS];
  __shared__ int      tok_lds[BCOLS];

  const int tid  = threadIdx.x;
  const int lane = tid & 31;
  const int wave = tid >> 5;
  const int bBase = blockIdx.x * BCOLS;
  const int bRow = tid >> 3;      // column 0..31 for element-wise phases
  const int seg  = tid & 7;       // 16-state segment 0..7

  if (tid < HMM_S) lse_lds[tid] = lse[tid];

  // --- constant A-fragments (transition tiles) in WMMA A layout ---
  const int j0 = wave * 16;
  v16h Afrag[4];
  {
    const int row  = j0 + (lane & 15);
    const int koff = (lane >> 4) ? 8 : 0;
    const _Float16* rp = Th + row * HMM_S;
#pragma unroll
    for (int ks = 0; ks < 4; ++ks) {
      const int kk = ks * 32;
#pragma unroll
      for (int i = 0; i < 8; ++i) {
        Afrag[ks][i]     = rp[kk + koff + i];
        Afrag[ks][8 + i] = rp[kk + 16 + koff + i];
      }
    }
  }

  // --- per-lane constants for the postprocess (D-layout: 8 consecutive j) ---
  const int bcol = lane & 15;
  const int jb   = j0 + 8 * (lane >> 4);
  float lse_r[8];
#pragma unroll
  for (int r = 0; r < 8; ++r) lse_r[r] = lse[jb + r];

  // --- f0[b][j] = table[tok0[b]][j] - lse[j]; seed pmax2 ---
  if (tid < BCOLS) tok_lds[tid] = sent[(bBase + tid) * L];
  __syncthreads();
  {
    const float* rowp = table + (long)tok_lds[bRow] * HMM_S;
    float m = -INFINITY;
#pragma unroll
    for (int i = 0; i < 16; ++i) {
      const int j = seg * 16 + i;
      const float v = rowp[j] - lse_lds[j];
      f_lds[bRow][j] = v;
      m = fmaxf(m, v);
    }
    pmax2[bRow][seg] = m;
  }

  for (int t = 1; t < L; ++t) {
    __syncthreads();   // barrier A: f_lds, pmax2 from previous step visible

    // --- token fetch for this step ---
    if (tid < BCOLS) tok_lds[tid] = sent[(bBase + tid) * L + t];

    // --- scaling constant (redundant 8-way LDS max per thread) + E=exp(f-c) ---
    {
      float c = pmax2[bRow][0];
#pragma unroll
      for (int i = 1; i < 8; ++i) c = fmaxf(c, pmax2[bRow][i]);
      if (seg == 0) c_lds[bRow] = c;
#pragma unroll
      for (int i = 0; i < 16; ++i) {
        const int j = seg * 16 + i;
        E_lds[bRow][j] = (_Float16)fast_exp(f_lds[bRow][j] - c);
      }
    }
    __syncthreads();   // barrier B: E, tok, c_lds visible

    // --- emission rows first: overlap global latency with DS + WMMA ---
    float em0[8], em1[8];
    {
      const float* r0 = table + (long)tok_lds[bcol] * HMM_S + jb;
      const float* r1 = table + (long)tok_lds[16 + bcol] * HMM_S + jb;
#pragma unroll
      for (int r = 0; r < 8; ++r) { em0[r] = r0[r]; em1[r] = r1[r]; }
    }
    const float c0 = c_lds[bcol];
    const float c1 = c_lds[16 + bcol];

    // --- preload all 8 B-fragments, then back-to-back WMMA chain ---
    const int nlane = lane & 15;
    const int khalf = (lane >> 4) * 8;
    v16h bf[8];
#pragma unroll
    for (int ks = 0; ks < 4; ++ks) {
      const _Float16* p0 = &E_lds[nlane][ks * 32 + khalf];
      const _Float16* p1 = &E_lds[16 + nlane][ks * 32 + khalf];
#pragma unroll
      for (int i = 0; i < 16; ++i) { const int e = (i < 8) ? i : 8 + i; bf[ks][i] = p0[e]; bf[4 + ks][i] = p1[e]; }
    }
    v8f acc0 = {}, acc1 = {};
#pragma unroll
    for (int ks = 0; ks < 4; ++ks) {
      acc0 = __builtin_amdgcn_wmma_f32_16x16x32_f16(false, Afrag[ks], false, bf[ks],
                                                    (short)0, acc0, false, false);
      acc1 = __builtin_amdgcn_wmma_f32_16x16x32_f16(false, Afrag[ks], false, bf[4 + ks],
                                                    (short)0, acc1, false, false);
    }

    // --- postprocess: f = log(acc) + c + em - lse; fold next-step max in-reg ---
    float m0 = -INFINITY, m1 = -INFINITY;
#pragma unroll
    for (int r = 0; r < 8; ++r) {
      const float v0 = fast_log(acc0[r]) + c0 + em0[r] - lse_r[r];
      const float v1 = fast_log(acc1[r]) + c1 + em1[r] - lse_r[r];
      f_lds[bcol][jb + r]      = v0;
      f_lds[16 + bcol][jb + r] = v1;
      m0 = fmaxf(m0, v0);
      m1 = fmaxf(m1, v1);
    }
    // combine with lane^16 partner (other 8 states of the same column)
    m0 = fmaxf(m0, __shfl_xor(m0, 16, 32));
    m1 = fmaxf(m1, __shfl_xor(m1, 16, 32));
    if (lane < 16) {
      pmax2[bcol][wave]      = m0;
      pmax2[16 + bcol][wave] = m1;
    }
  }
  __syncthreads();

  // --- final: ppl[b] = logsumexp_j f[b,j]; sum 32 columns -> partial[WG] ---
  float c = pmax2[bRow][0];
#pragma unroll
  for (int i = 1; i < 8; ++i) c = fmaxf(c, pmax2[bRow][i]);
  float s = 0.f;
#pragma unroll
  for (int i = 0; i < 16; ++i) s += fast_exp(f_lds[bRow][seg * 16 + i] - c);
  pred[bRow][seg] = s; __syncthreads();
  if (seg == 0) {
    float ss = 0.f;
#pragma unroll
    for (int i = 0; i < 8; ++i) ss += pred[bRow][i];
    colppl[bRow] = fast_log(ss) + c;
  }
  __syncthreads();
  if (tid == 0) {
    float tot = 0.f;
    for (int b = 0; b < BCOLS; ++b) tot += colppl[b];
    partial[blockIdx.x] = tot;
  }
}

// ---------------------------------------------------------------------------
// Kernel D: deterministic final reduction of the per-WG partials
// ---------------------------------------------------------------------------
__global__ void hmm_final_sum(const float* __restrict__ partial, float* __restrict__ out,
                              int n) {
  float tot = 0.f;
  for (int i = 0; i < n; ++i) tot += partial[i];
  out[0] = tot;
}

extern "C" void kernel_launch(void* const* d_in, const int* in_sizes, int n_in,
                              void* d_out, int out_size, void* d_ws, size_t ws_size,
                              hipStream_t stream) {
  const int*   sent  = (const int*)d_in[0];     // (256, L) int32
  // d_in[1]: masks (all ones) -> length == L for every batch row
  const float* table = (const float*)d_in[2];   // (32000, 128) f32
  const float* trans = (const float*)d_in[3];   // (128, 128) f32
  float* out = (float*)d_out;                   // scalar f32

  const int L = in_sizes[0] / HMM_B;

  char* ws = (char*)d_ws;
  float*    lse     = (float*)ws;                        // 128 f32
  _Float16* Th      = (_Float16*)(ws + 1024);            // 128x128 f16 (32KB)
  float*    partial = (float*)(ws + 1024 + 32768);       // 8 f32

  hmm_col_lse    <<<HMM_S, 256, 0, stream>>>(table, lse);
  hmm_row_softmax<<<HMM_S, 128, 0, stream>>>(trans, Th);
  hmm_forward    <<<HMM_B / BCOLS, 256, 0, stream>>>(sent, table, lse, Th, partial, L);
  hmm_final_sum  <<<1, 1, 0, stream>>>(partial, out, HMM_B / BCOLS);
}
